// DistanceModel_86320252715188
// MI455X (gfx1250) — matrix operand order repaired and checked by reading, not compiled, not run on hardware
//
#include <hip/hip_runtime.h>
#include <hip/hip_bf16.h>

typedef __attribute__((ext_vector_type(16))) _Float16 v16h;
typedef __attribute__((ext_vector_type(8)))  float    v8f;

#define IN_F   128
#define HID_F  256
#define OUT_F  128
#define NEG_SLOPE 0.01f

__device__ __forceinline__ v8f wmma16x16x32(v16h a, v16h b, v8f c) {
  // D = A(16x32 f16) * B(32x16 f16) + C(16x16 f32)
  return __builtin_amdgcn_wmma_f32_16x16x32_f16(false, a, false, b, (short)0, c,
                                                false, false);
}

// Packed-fragment load: 32 contiguous bytes per lane (2x global_load_b128).
__device__ __forceinline__ v16h load_frag(const _Float16* __restrict__ buf,
                                          long long tile) {
  int lane = threadIdx.x & 31;
  return *(const v16h*)(buf + tile * 512 + lane * 16);
}

// =====================================================================
// Prepack kernels: fp32 matrices -> f16 WMMA fragments (lane-major, 512
// halves per 16x32 / 32x16 tile; GEMM loops then load 32B contiguous).

// Weight [K x N] row-major. tile id = nt*(K/32) + kt  (k-tiles contiguous)
__global__ void k_pack_w(const float* __restrict__ W, _Float16* __restrict__ P,
                         int K, int N) {
  int KT = K >> 5;
  int t = blockIdx.x;
  int nt = t / KT, kt = t - nt * KT;
  int lane = threadIdx.x & 31;
  int c  = nt * 16 + (lane & 15);
  int kk = kt * 32 + ((lane & 16) ? 8 : 0);
  v16h b;
#pragma unroll
  for (int j = 0; j < 16; ++j) b[j] = (_Float16)W[(size_t)(kk + j + (j >= 8 ? 8 : 0)) * N + c];
  *(v16h*)(P + (size_t)t * 512 + lane * 16) = b;
}

// Activation [n x K] row-major. tile id = rt*(K/32) + kt
// A-fragment lane layout: L<16: row=L,    K=[0..7],[16..23]
//                         L>=16: row=L-16, K=[8..15],[24..31]
__global__ void k_pack_a(const float* __restrict__ A, _Float16* __restrict__ P,
                         int n, int K) {
  int KT = K >> 5;
  int t = blockIdx.x;
  int rt = t / KT, kt = t - rt * KT;
  int lane = threadIdx.x & 31;
  int r = rt * 16 + (lane & 15);
  if (r >= n) r = n - 1;
  const float* p = A + (size_t)r * K + kt * 32 + ((lane & 16) ? 8 : 0);
  v16h a;
#pragma unroll
  for (int j = 0; j < 8; ++j) {
    a[j]     = (_Float16)p[j];
    a[8 + j] = (_Float16)p[16 + j];
  }
  *(v16h*)(P + (size_t)t * 512 + lane * 16) = a;
}

// Same, row scaled by 1/max(deg,1): the segment-mean divide is fused into
// the one-time pack instead of being redone for every column tile.
__global__ void k_pack_a_scaled(const float* __restrict__ A,
                                const float* __restrict__ deg,
                                _Float16* __restrict__ P, int n, int K) {
  int KT = K >> 5;
  int t = blockIdx.x;
  int rt = t / KT, kt = t - rt * KT;
  int lane = threadIdx.x & 31;
  int r = rt * 16 + (lane & 15);
  if (r >= n) r = n - 1;
  float s = 1.0f / fmaxf(deg[r], 1.0f);
  const float* p = A + (size_t)r * K + kt * 32 + ((lane & 16) ? 8 : 0);
  v16h a;
#pragma unroll
  for (int j = 0; j < 8; ++j) {
    a[j]     = (_Float16)(p[j] * s);
    a[8 + j] = (_Float16)(p[16 + j] * s);
  }
  *(v16h*)(P + (size_t)t * 512 + lane * 16) = a;
}

// =====================================================================

__global__ void k_fill_zero(float* p, long long n) {
  long long i = (long long)blockIdx.x * blockDim.x + threadIdx.x;
  long long stride = (long long)gridDim.x * blockDim.x;
  for (; i < n; i += stride) p[i] = 0.0f;
}

// segment-sum feat[src[e]] into accum[dst[e]]; optionally count degree
__global__ void k_scatter_add(const float* __restrict__ feat,
                              const int* __restrict__ src,
                              const int* __restrict__ dst,
                              float* accum, float* deg, int nE, int F) {
  int per = F >> 2;
  long long t = (long long)blockIdx.x * blockDim.x + threadIdx.x;
  if (t >= (long long)nE * per) return;
  int e  = (int)(t / per);
  int f4 = (int)(t % per) << 2;
  int s = src[e], d = dst[e];
  const float* fp = feat + (size_t)s * F + f4;
  float* ap = accum + (size_t)d * F + f4;
  atomicAdd(ap + 0, fp[0]);
  atomicAdd(ap + 1, fp[1]);
  atomicAdd(ap + 2, fp[2]);
  atomicAdd(ap + 3, fp[3]);
  if (deg != nullptr && f4 == 0) atomicAdd(deg + d, 1.0f);
}

// Fully-packed dual GEMM: out = [relu]( self@Ws + neigh@Wn + bias ).
// One wave per 16x16 output tile; inner loop = 4 frag loads + 2 wmma.
__global__ void k_gemm_pp(const _Float16* __restrict__ selfP,
                          const _Float16* __restrict__ neighP,
                          const _Float16* __restrict__ WsP,
                          const _Float16* __restrict__ WnP,
                          const float* __restrict__ bias, float* __restrict__ out,
                          int n, int KT, int Nout, int do_relu) {
  int rt = blockIdx.x, nt = blockIdx.y;
  int lane = threadIdx.x & 31;
  v8f acc = {};
  for (int kt = 0; kt < KT; ++kt) {
    acc = wmma16x16x32(load_frag(selfP,  (long long)rt * KT + kt),
                       load_frag(WsP,    (long long)nt * KT + kt), acc);
    acc = wmma16x16x32(load_frag(neighP, (long long)rt * KT + kt),
                       load_frag(WnP,    (long long)nt * KT + kt), acc);
  }
  int col = nt * 16 + (lane & 15);
  float bv = bias[col];
  int rbase = rt * 16 + ((lane & 16) ? 8 : 0);
  float* op = out + (size_t)rbase * Nout + col;
  if (rt * 16 + 16 <= n) {        // full tile: branch-free stores
#pragma unroll
    for (int r = 0; r < 8; ++r) {
      float v = acc[r] + bv;
      if (do_relu) v = fmaxf(v, 0.0f);
      op[(size_t)r * Nout] = v;
    }
  } else {
    for (int r = 0; r < 8; ++r) {
      if (rbase + r < n) {
        float v = acc[r] + bv;
        if (do_relu) v = fmaxf(v, 0.0f);
        op[(size_t)r * Nout] = v;
      }
    }
  }
}

// f_ni = x@W_ni, f_nj = x@W_nj  (shared packed-A fragment, two accumulators)
__global__ void k_ninj(const _Float16* __restrict__ xP,
                       const _Float16* __restrict__ WniP,
                       const _Float16* __restrict__ WnjP,
                       float* __restrict__ fni, float* __restrict__ fnj, int n) {
  int rt = blockIdx.x, nt = blockIdx.y;
  int lane = threadIdx.x & 31;
  const int KT = IN_F >> 5;  // 4
  v8f ai = {}, aj = {};
#pragma unroll
  for (int kt = 0; kt < KT; ++kt) {
    v16h a = load_frag(xP, (long long)rt * KT + kt);
    ai = wmma16x16x32(a, load_frag(WniP, (long long)nt * KT + kt), ai);
    aj = wmma16x16x32(a, load_frag(WnjP, (long long)nt * KT + kt), aj);
  }
  int col = nt * 16 + (lane & 15);
  int rbase = rt * 16 + ((lane & 16) ? 8 : 0);
  float* pi = fni + (size_t)rbase * OUT_F + col;
  float* pj = fnj + (size_t)rbase * OUT_F + col;
  if (rt * 16 + 16 <= n) {
#pragma unroll
    for (int r = 0; r < 8; ++r) {
      pi[(size_t)r * OUT_F] = ai[r];
      pj[(size_t)r * OUT_F] = aj[r];
    }
  } else {
    for (int r = 0; r < 8; ++r) {
      if (rbase + r < n) {
        pi[(size_t)r * OUT_F] = ai[r];
        pj[(size_t)r * OUT_F] = aj[r];
      }
    }
  }
}

// Fused edge kernel. 256 threads = 8 waves; wave w owns cols [16w,16w+16).
// Edge tile (16x128 fp32) staged once into LDS as f16 fragments; then
// f_out = leakyrelu(e@W_fij + f_ni[src] + f_nj[dst] + be) kept in LDS;
// score = concat(h2[src], h2[dst], f_out) @ Wp + bp.
__global__ void k_edge(const float* __restrict__ efeat,
                       const int* __restrict__ src, const int* __restrict__ dst,
                       const _Float16* __restrict__ WfP, const float* __restrict__ be,
                       const float* __restrict__ fni, const float* __restrict__ fnj,
                       const float* __restrict__ h2,
                       const float* __restrict__ Wp, const float* __restrict__ bp,
                       float* __restrict__ out, int nE) {
  __shared__ __align__(32) _Float16 aLDS[4 * 512];  // 4 k-tiles of A fragments
  __shared__ float fout[16][132];
  __shared__ int s_src[16], s_dst[16];

  int m0   = blockIdx.x * 16;
  int wave = threadIdx.x >> 5;
  int lane = threadIdx.x & 31;

  if (threadIdx.x < 16) {
    int ge = m0 + threadIdx.x;
    if (ge >= nE) ge = nE - 1;
    s_src[threadIdx.x] = src[ge];
    s_dst[threadIdx.x] = dst[ge];
  }

  // Cooperative stage: fp32 edge tile -> f16 fragments in LDS.
  // 256 threads x 8 halves: tt -> (ktile, lane2, group)
  {
    int tt = threadIdx.x;
    int ktile = tt >> 6;
    int lane2 = (tt >> 1) & 31;
    int g     = tt & 1;
    int row = m0 + (lane2 & 15);
    if (row >= nE) row = nE - 1;
    int kbase = ktile * 32 + ((lane2 & 16) ? 8 : 0) + (g ? 16 : 0);
    const float* p = efeat + (size_t)row * IN_F + kbase;
    _Float16* dp = aLDS + ktile * 512 + lane2 * 16 + g * 8;
#pragma unroll
    for (int j = 0; j < 8; ++j) dp[j] = (_Float16)p[j];
  }
  __syncthreads();

  v8f acc = {};
  const int KT = IN_F >> 5;  // 4
#pragma unroll
  for (int kt = 0; kt < KT; ++kt) {
    v16h a = *(const v16h*)(aLDS + kt * 512 + lane * 16);      // 2x ds_load_b128
    v16h b = load_frag(WfP, (long long)wave * KT + kt);        // 2x global b128
    acc = wmma16x16x32(a, b, acc);
  }

  int col = wave * 16 + (lane & 15);
  float bv = be[col];
#pragma unroll
  for (int r = 0; r < 8; ++r) {
    int mr = r + ((lane & 16) ? 8 : 0);         // local edge row 0..15
    int sI = s_src[mr], dI = s_dst[mr];
    float v = acc[r] + bv
            + fni[(size_t)sI * OUT_F + col]     // L2-resident gathers
            + fnj[(size_t)dI * OUT_F + col];
    v = (v >= 0.0f) ? v : NEG_SLOPE * v;        // leaky_relu
    fout[mr][col] = v;
  }
  __syncthreads();

  // Predictor: 48 threads, one (edge, k) each; Wp is [384 x 3]
  if (threadIdx.x < 48) {
    int le = threadIdx.x / 3, k = threadIdx.x % 3;
    int ge = m0 + le;
    if (ge < nE) {
      const float* hs = h2 + (size_t)s_src[le] * OUT_F;
      const float* hd = h2 + (size_t)s_dst[le] * OUT_F;
      float a2 = bp[k];
#pragma unroll 4
      for (int d = 0; d < OUT_F; ++d) {
        a2 += hs[d]       * Wp[(size_t)d * 3 + k]
            + hd[d]       * Wp[(size_t)(OUT_F + d) * 3 + k]
            + fout[le][d] * Wp[(size_t)(2 * OUT_F + d) * 3 + k];
      }
      out[(size_t)ge * 3 + k] = a2;
    }
  }
}

// =====================================================================

extern "C" void kernel_launch(void* const* d_in, const int* in_sizes, int n_in,
                              void* d_out, int out_size, void* d_ws, size_t ws_size,
                              hipStream_t stream) {
  const float* x    = (const float*)d_in[0];
  const float* e    = (const float*)d_in[1];
  const int*   src  = (const int*)d_in[2];
  const int*   dst  = (const int*)d_in[3];
  const float* Ws1  = (const float*)d_in[4];
  const float* Wn1  = (const float*)d_in[5];
  const float* b1   = (const float*)d_in[6];
  const float* Ws2  = (const float*)d_in[7];
  const float* Wn2  = (const float*)d_in[8];
  const float* b2   = (const float*)d_in[9];
  const float* Wni  = (const float*)d_in[10];
  const float* Wnj  = (const float*)d_in[11];
  const float* Wfij = (const float*)d_in[12];
  // d_in[13] W_node, d_in[14] attn: dead code w.r.t. the returned score
  const float* be   = (const float*)d_in[15];
  const float* Wp   = (const float*)d_in[16];
  const float* bp   = (const float*)d_in[17];
  float* out = (float*)d_out;

  const int n  = in_sizes[0] / IN_F;   // 50000
  const int nE = in_sizes[2];          // 400000
  const int mt = (n + 15) / 16;
  const int et = (nE + 15) / 16;

  // ---- workspace layout: fp32 region, then f16 fragment region ----
  float* ws     = (float*)d_ws;
  float* deg    = ws;                          // n
  float* accum1 = deg    + n;                  // n*128
  float* accum2 = accum1 + (size_t)n * IN_F;   // n*256
  float* h      = accum2 + (size_t)n * HID_F;  // n*256
  float* h2     = h      + (size_t)n * HID_F;  // n*128
  float* fni    = h2     + (size_t)n * OUT_F;  // n*128
  float* fnj    = fni    + (size_t)n * OUT_F;  // n*128
  uintptr_t up = ((uintptr_t)(fnj + (size_t)n * OUT_F) + 31) & ~(uintptr_t)31;
  _Float16* xP   = (_Float16*)up;              // mt*4 tiles (x, f16 frags)
  _Float16* n1P  = xP   + (size_t)mt * 4 * 512;  // neigh1 (accum1/deg)
  _Float16* hP   = n1P  + (size_t)mt * 4 * 512;  // h,      mt*8 tiles
  _Float16* n2P  = hP   + (size_t)mt * 8 * 512;  // neigh2 (accum2/deg)
  _Float16* pWs1 = n2P  + (size_t)mt * 8 * 512;
  _Float16* pWn1 = pWs1 + 64 * 512;            // Ws1: 16 n-tiles * 4 k-tiles
  _Float16* pWs2 = pWn1 + 64 * 512;
  _Float16* pWn2 = pWs2 + 64 * 512;            // Ws2: 8 n-tiles * 8 k-tiles
  _Float16* pWni = pWn2 + 64 * 512;
  _Float16* pWnj = pWni + 32 * 512;            // Wni: 8 n-tiles * 4 k-tiles
  _Float16* pWfj = pWnj + 32 * 512;
  (void)ws_size; (void)n_in; (void)out_size;

  // 0) prepack weights + x into f16 fragments
  k_pack_w<<<64, 32, 0, stream>>>(Ws1, pWs1, IN_F,  HID_F);
  k_pack_w<<<64, 32, 0, stream>>>(Wn1, pWn1, IN_F,  HID_F);
  k_pack_w<<<64, 32, 0, stream>>>(Ws2, pWs2, HID_F, OUT_F);
  k_pack_w<<<64, 32, 0, stream>>>(Wn2, pWn2, HID_F, OUT_F);
  k_pack_w<<<32, 32, 0, stream>>>(Wni, pWni, IN_F,  OUT_F);
  k_pack_w<<<32, 32, 0, stream>>>(Wnj, pWnj, IN_F,  OUT_F);
  k_pack_w<<<32, 32, 0, stream>>>(Wfij, pWfj, IN_F, OUT_F);
  k_pack_a<<<mt * 4, 32, 0, stream>>>(x, xP, n, IN_F);

  // 1) zero deg + accum1 + accum2 (contiguous)
  k_fill_zero<<<2048, 256, 0, stream>>>(deg, (long long)n * (1 + IN_F + HID_F));

  // 2) segment sum of x[src] by dst, plus degree
  {
    long long t = (long long)nE * (IN_F / 4);
    k_scatter_add<<<(int)((t + 255) / 256), 256, 0, stream>>>(x, src, dst,
                                                              accum1, deg, nE, IN_F);
  }
  // 3) pack neigh1 = accum1/deg, then SAGE1: h = relu(x@Ws1 + n1@Wn1 + b1)
  k_pack_a_scaled<<<mt * 4, 32, 0, stream>>>(accum1, deg, n1P, n, IN_F);
  k_gemm_pp<<<dim3(mt, HID_F / 16), 32, 0, stream>>>(xP, n1P, pWs1, pWn1, b1,
                                                     h, n, IN_F / 32, HID_F, 1);
  // 4) segment sum of h[src] by dst
  {
    long long t = (long long)nE * (HID_F / 4);
    k_scatter_add<<<(int)((t + 255) / 256), 256, 0, stream>>>(h, src, dst,
                                                              accum2, nullptr, nE, HID_F);
  }
  // 5) pack h and neigh2, then SAGE2: h2 = h@Ws2 + n2@Wn2 + b2
  k_pack_a<<<mt * 8, 32, 0, stream>>>(h, hP, n, HID_F);
  k_pack_a_scaled<<<mt * 8, 32, 0, stream>>>(accum2, deg, n2P, n, HID_F);
  k_gemm_pp<<<dim3(mt, OUT_F / 16), 32, 0, stream>>>(hP, n2P, pWs2, pWn2, b2,
                                                     h2, n, HID_F / 32, OUT_F, 0);
  // 6) f_ni / f_nj node projections
  k_ninj<<<dim3(mt, OUT_F / 16), 32, 0, stream>>>(xP, pWni, pWnj, fni, fnj, n);
  // 7) fused edge GEMM + leaky_relu + predictor
  k_edge<<<et, 256, 0, stream>>>(e, src, dst, pWfj, be, fni, fnj, h2, Wp, bp,
                                 out, nE);
}
